// PhaseAttentionLM_3650722201706
// MI455X (gfx1250) — compile-verified, matrix operand order checked by reading, not run on hardware
//
#include <hip/hip_runtime.h>
#include <hip/hip_bf16.h>
#include <math.h>

typedef __attribute__((ext_vector_type(16))) _Float16 v16h;
typedef __attribute__((ext_vector_type(8)))  _Float16 v8h;
typedef __attribute__((ext_vector_type(8)))  float    v8f;
typedef __attribute__((ext_vector_type(4)))  int      v4i;

__device__ __forceinline__ v8f wmma16(v16h a, v16h b, v8f c) {
  // D = A(16x32 f16) * B(32x16 f16) + C(16x16 f32)
  return __builtin_amdgcn_wmma_f32_16x16x32_f16(false, a, false, b, (short)0, c,
                                                false, false);
}

// K-offset within a 32-wide fragment step for VGPR pair v (0..7), half h (0..1).
// Elements a[0..7] cover K = h*8 + 0..7 (contiguous), a[8..15] cover 16+h*8+0..7.
__device__ __forceinline__ int frag_k(int v, int h) {
  return ((v < 4) ? 0 : 16) + h * 8 + (v & 3) * 2;
}

// --------------------------------------------------------------------------
// CDNA5 async global->LDS copy (ASYNCcnt path), guarded so we always compile.
// Probe result: builtin takes (int4 AS1*, int4 AS3*, imm offset, imm cpol).
// --------------------------------------------------------------------------
#if defined(__HIP_DEVICE_COMPILE__) && \
    __has_builtin(__builtin_amdgcn_global_load_async_to_lds_b128)
#define ASYNC_LDS 1
typedef __attribute__((address_space(1))) v4i* g_v4i_p;
typedef __attribute__((address_space(3))) v4i* l_v4i_p;
#endif

__device__ __forceinline__ void cp16_g2l(const void* g, void* l) {
#ifdef ASYNC_LDS
  __builtin_amdgcn_global_load_async_to_lds_b128((g_v4i_p)(void*)g,
                                                 (l_v4i_p)l, 0, 0);
#else
  *(float4*)l = *(const float4*)g;
#endif
}

template <int N>
__device__ __forceinline__ void wait_async_n() {
#ifdef ASYNC_LDS
#if __has_builtin(__builtin_amdgcn_s_wait_asynccnt)
  __builtin_amdgcn_s_wait_asynccnt(N);
#else
  asm volatile("s_wait_asynccnt %0" ::"i"(N) : "memory");
#endif
#endif
}

// ---------------------------------------------------------------------------
// f32 -> f16 elementwise convert (one-time weight shadow copies)
// ---------------------------------------------------------------------------
__global__ void cvt_f16_kernel(const float* __restrict__ src,
                               _Float16* __restrict__ dst, size_t n) {
  size_t i = (size_t)blockIdx.x * blockDim.x + threadIdx.x;
  if (i < n) dst[i] = (_Float16)src[i];
}

// ---------------------------------------------------------------------------
// Embedding gather: x f32 + f16 shadow
// ---------------------------------------------------------------------------
__global__ void embed_kernel(const int* __restrict__ idx,
                             const float* __restrict__ TE,
                             const float* __restrict__ PE,
                             float* __restrict__ X, _Float16* __restrict__ Xh,
                             int BL, int L, int D) {
  size_t i = (size_t)blockIdx.x * blockDim.x + threadIdx.x;
  size_t total = (size_t)BL * D;
  if (i >= total) return;
  int row = (int)(i / D);
  int d   = (int)(i % D);
  int l   = row % L;
  const float v = TE[(size_t)idx[row] * D + d] + PE[(size_t)l * D + d];
  X[i]  = v;
  Xh[i] = (_Float16)v;
}

// ---------------------------------------------------------------------------
// Skinny GEMM (N small, e.g. 64): one wave per 16x16 tile, f16 operands.
// act: 0 none, 1 elu(x)+1
// ---------------------------------------------------------------------------
__global__ void wmma_gemm_small(const _Float16* __restrict__ A,
                                const _Float16* __restrict__ B,
                                float* __restrict__ C, int M, int N, int K,
                                int act) {
  const int tilesN = N >> 4;
  const int tilesM = M >> 4;
  const long wave = (long)blockIdx.x * (blockDim.x >> 5) + (threadIdx.x >> 5);
  if (wave >= (long)tilesM * tilesN) return;
  const int tm = (int)(wave / tilesN);
  const int tn = (int)(wave % tilesN);
  const int lane = threadIdx.x & 31;
  const int mn = lane & 15;
  const int h  = lane >> 4;

  const _Float16* Arow = A + (size_t)(tm * 16 + mn) * K;
  const _Float16* Bcol = B + (size_t)(tn * 16 + mn);

  v8f acc = {};
  for (int k0 = 0; k0 < K; k0 += 32) {
    const v8h alo = *(const v8h*)(Arow + k0 + h * 8);
    const v8h ahi = *(const v8h*)(Arow + k0 + 16 + h * 8);
    const v16h a = __builtin_shufflevector(alo, ahi, 0, 1, 2, 3, 4, 5, 6, 7, 8,
                                           9, 10, 11, 12, 13, 14, 15);
    v16h b;
#pragma unroll
    for (int c = 0; c < 2; ++c)
#pragma unroll
      for (int e = 0; e < 8; ++e)
        b[c * 8 + e] = Bcol[(size_t)(k0 + c * 16 + h * 8 + e) * N];
    acc = wmma16(a, b, acc);
  }

#pragma unroll
  for (int v = 0; v < 8; ++v) {
    const int mo = tm * 16 + v + 8 * h;
    float f = acc[v];
    if (act == 1) f = (f > 0.f) ? (f + 1.f) : __expf(f);  // elu+1
    C[(size_t)mo * N + tn * 16 + mn] = f;
  }
}

// ---------------------------------------------------------------------------
// Tiled GEMM: 128x128x32 macro-tile per block (8 waves), raw-f16 LDS tiles,
// DOUBLE-BUFFERED async global->LDS fills (overlap copy with WMMA compute).
// Requires M%128==0, N%128==0, K%32==0.
// Each wave: 64x32 = 4x2 WMMA tiles. A-fragments are two ds_load_b128's.
// Output: f32 (bias/act/residual) or f16 (C16).
// ---------------------------------------------------------------------------
#define TBM 128
#define TBN 128
#define TBK 32
#define LDAH 40    // padded f16 stride for A tile (16B-aligned rows)
#define LDBH 136   // padded f16 stride for B tile (16B-aligned rows)

__global__ __launch_bounds__(256) void wmma_gemm_tiled(
    const _Float16* __restrict__ A, const _Float16* __restrict__ B,
    const float* __restrict__ bias, const float* __restrict__ residual,
    float* __restrict__ C, _Float16* __restrict__ C16, int M, int N, int K,
    int act) {
  __shared__ _Float16 lA[2][TBM * LDAH];  // [buf][row][k]
  __shared__ _Float16 lB[2][TBK * LDBH];  // [buf][k][n]

  const int nBlocks = N / TBN;
  const int bm = blockIdx.x / nBlocks;
  const int bn = blockIdx.x % nBlocks;
  const int tid = threadIdx.x;
  const int wave = tid >> 5, lane = tid & 31;
  const int waveM = wave >> 2;     // 0..1 -> 64 rows each
  const int waveN = wave & 3;      // 0..3 -> 32 cols each
  const int mn = lane & 15, h = lane >> 4;

  const _Float16* Ablk = A + (size_t)bm * TBM * K;
  const _Float16* Bblk = B + (size_t)bn * TBN;

  // Each call issues 4 async (or plain) 16B copies per thread.
  auto fill = [&](int buf, int k0) {
#pragma unroll
    for (int i = 0; i < 2; ++i) {
      const int s = tid + 256 * i;
      {
        const int row = s >> 2, sc = (s & 3) * 8;    // A: 4 x 8-half segs / row
        cp16_g2l(Ablk + (size_t)row * K + k0 + sc, &lA[buf][row * LDAH + sc]);
      }
      {
        const int row = s >> 4, sc = (s & 15) * 8;   // B: 16 x 8-half segs / row
        cp16_g2l(Bblk + (size_t)(k0 + row) * N + sc, &lB[buf][row * LDBH + sc]);
      }
    }
  };

  v8f acc[4][2] = {};
  const int nIter = K / TBK;

  fill(0, 0);
  for (int it = 0; it < nIter; ++it) {
    const int cur = it & 1;
    const bool more = (it + 1) < nIter;
    // Stream next K-slab into the other buffer while we compute this one.
    if (more) fill(cur ^ 1, (it + 1) * TBK);
    if (it + 2 < nIter) {  // L2 prefetch two slabs ahead
      const int s = tid;
      const int kp = (it + 2) * TBK;
      __builtin_prefetch(Ablk + (size_t)(s >> 2) * K + kp + (s & 3) * 8, 0, 1);
      __builtin_prefetch(Bblk + (size_t)(kp + (s >> 4)) * N + (s & 15) * 8, 0, 1);
    }
    // Drain only the previous slab's 4 copies (async completes in order).
    if (more) wait_async_n<4>(); else wait_async_n<0>();
    __syncthreads();

    // ---- fragments straight from f16 LDS ----
    v16h af[4], bf[2];
#pragma unroll
    for (int i = 0; i < 4; ++i) {
      const _Float16* Ar = &lA[cur][(waveM * 64 + i * 16 + mn) * LDAH];
      const v8h lo = *(const v8h*)(Ar + h * 8);
      const v8h hi = *(const v8h*)(Ar + 16 + h * 8);
      af[i] = __builtin_shufflevector(lo, hi, 0, 1, 2, 3, 4, 5, 6, 7, 8, 9, 10,
                                      11, 12, 13, 14, 15);
    }
#pragma unroll
    for (int j = 0; j < 2; ++j) {
      const int col = waveN * 32 + j * 16 + mn;
#pragma unroll
      for (int c = 0; c < 2; ++c)
#pragma unroll
        for (int e = 0; e < 8; ++e)
          bf[j][c * 8 + e] = lB[cur][(c * 16 + h * 8 + e) * LDBH + col];
    }

#pragma unroll
    for (int i = 0; i < 4; ++i)
#pragma unroll
      for (int j = 0; j < 2; ++j)
        acc[i][j] = wmma16(af[i], bf[j], acc[i][j]);

    __syncthreads();  // protect buffer reuse two iterations out
  }

  // ---- epilogue ----
#pragma unroll
  for (int i = 0; i < 4; ++i) {
    const int moB = bm * TBM + waveM * 64 + i * 16;
#pragma unroll
    for (int j = 0; j < 2; ++j) {
      const int no = bn * TBN + waveN * 32 + j * 16 + mn;
      const float bv = bias ? bias[no] : 0.f;
#pragma unroll
      for (int v = 0; v < 8; ++v) {
        const int mo = moB + v + 8 * h;
        float f = acc[i][j][v] + bv;
        if (act == 2) f = 0.5f * f * (1.f + erff(f * 0.70710678f));  // gelu
        const size_t o = (size_t)mo * N + no;
        if (C16) {
          C16[o] = (_Float16)f;
        } else {
          if (residual) f += residual[o];
          C[o] = f;
        }
      }
    }
  }
}

// ---------------------------------------------------------------------------
// Per-chunk KV state: KVc[b,c] = K_c^T @ V_c (64x64), Ksumc[b,c] = sum_t K_t
// ---------------------------------------------------------------------------
__global__ void chunk_kv_kernel(const float* __restrict__ Km,
                                const float* __restrict__ Vm,
                                float* __restrict__ KVc,
                                float* __restrict__ Ksumc) {
  const int bc = blockIdx.x;
  const float* Kc = Km + (size_t)bc * 64 * 64;
  const float* Vc = Vm + (size_t)bc * 64 * 64;
  float* outKV = KVc + (size_t)bc * 64 * 64;

  const int wave = threadIdx.x >> 5;
  const int tm = wave >> 2, tn = wave & 3;
  const int lane = threadIdx.x & 31;
  const int mn = lane & 15, h = lane >> 4;
  const int row = tm * 16 + mn;
  const int col = tn * 16 + mn;

  v8f acc = {};
  for (int k0 = 0; k0 < 64; k0 += 32) {
    v16h a, b;
#pragma unroll
    for (int v = 0; v < 8; ++v) {
      const int kk = k0 + frag_k(v, h);
      a[2 * v]     = (_Float16)Kc[kk * 64 + row];
      a[2 * v + 1] = (_Float16)Kc[(kk + 1) * 64 + row];
      b[2 * v]     = (_Float16)Vc[kk * 64 + col];
      b[2 * v + 1] = (_Float16)Vc[(kk + 1) * 64 + col];
    }
    acc = wmma16(a, b, acc);
  }
#pragma unroll
  for (int v = 0; v < 8; ++v)
    outKV[(tm * 16 + v + 8 * h) * 64 + col] = acc[v];

  if (threadIdx.x < 64) {
    float s = 0.f;
    for (int t = 0; t < 64; ++t) s += Kc[t * 64 + threadIdx.x];
    Ksumc[(size_t)bc * 64 + threadIdx.x] = s;
  }
}

// ---------------------------------------------------------------------------
// Exclusive prefix over chunks. One block per batch.
// ---------------------------------------------------------------------------
__global__ void prefix_kernel(const float* __restrict__ KVc,
                              const float* __restrict__ Ksumc,
                              float* __restrict__ KVp, float* __restrict__ Ksp,
                              int NC) {
  const int b = blockIdx.x;
  for (int e = threadIdx.x; e < 64 * 64; e += blockDim.x) {
    float s = 0.f;
    for (int c = 0; c < NC; ++c) {
      const size_t o = (size_t)(b * NC + c) * 4096 + e;
      KVp[o] = s;
      s += KVc[o];
    }
  }
  for (int e = threadIdx.x; e < 64; e += blockDim.x) {
    float s = 0.f;
    for (int c = 0; c < NC; ++c) {
      const size_t o = (size_t)(b * NC + c) * 64 + e;
      Ksp[o] = s;
      s += Ksumc[o];
    }
  }
}

// ---------------------------------------------------------------------------
// Intra+inter chunk attention (chunked causal linear attention). f16 output.
// ---------------------------------------------------------------------------
__global__ void attn_chunk_kernel(const float* __restrict__ Q,
                                  const float* __restrict__ Km,
                                  const float* __restrict__ V,
                                  const float* __restrict__ KVp,
                                  const float* __restrict__ Ksp,
                                  _Float16* __restrict__ Out) {
  __shared__ float S[64 * 65];
  __shared__ float den[64];
  const int bc = blockIdx.x;
  const float* Qc   = Q   + (size_t)bc * 4096;
  const float* Kc   = Km  + (size_t)bc * 4096;
  const float* Vc   = V   + (size_t)bc * 4096;
  const float* KVpc = KVp + (size_t)bc * 4096;
  const float* Kspc = Ksp + (size_t)bc * 64;
  _Float16* Oc = Out + (size_t)bc * 4096;

  const int wave = threadIdx.x >> 5;
  const int tm = wave >> 2, tn = wave & 3;
  const int lane = threadIdx.x & 31;
  const int mn = lane & 15, h = lane >> 4;

  // Phase 1: S tile = Q_c @ K_c^T
  {
    const float* Arow = Qc + (size_t)(tm * 16 + mn) * 64;
    const float* Brow = Kc + (size_t)(tn * 16 + mn) * 64;
    v8f acc = {};
    for (int k0 = 0; k0 < 64; k0 += 32) {
      v16h a, b;
#pragma unroll
      for (int v = 0; v < 8; ++v) {
        const int kk = k0 + frag_k(v, h);
        const float2 ta = *(const float2*)(Arow + kk);
        const float2 tb = *(const float2*)(Brow + kk);
        a[2 * v] = (_Float16)ta.x; a[2 * v + 1] = (_Float16)ta.y;
        b[2 * v] = (_Float16)tb.x; b[2 * v + 1] = (_Float16)tb.y;
      }
      acc = wmma16(a, b, acc);
    }
#pragma unroll
    for (int v = 0; v < 8; ++v) {
      const int t = tm * 16 + v + 8 * h;
      const int s = tn * 16 + mn;
      S[t * 65 + s] = (s <= t) ? acc[v] : 0.f;   // causal mask
    }
  }
  __syncthreads();

  // den[t] = rowsum(S) + Q_t . Ksum_prefix
  if (threadIdx.x < 64) {
    const int t = threadIdx.x;
    float s = 0.f;
    for (int ss = 0; ss <= t; ++ss) s += S[t * 65 + ss];
    const float* q = Qc + (size_t)t * 64;
    for (int k = 0; k < 64; ++k) s += q[k] * Kspc[k];
    den[t] = s + 1e-6f;
  }
  __syncthreads();

  // Phase 2: num = S @ V_c + Q_c @ KVprefix (same accumulator)
  v8f acc = {};
  {
    const float* Arow = &S[(tm * 16 + mn) * 65];
    const float* Bcol = Vc + tn * 16 + mn;
    for (int k0 = 0; k0 < 64; k0 += 32) {
      v16h a, b;
#pragma unroll
      for (int v = 0; v < 8; ++v) {
        const int kk = k0 + frag_k(v, h);
        a[2 * v]     = (_Float16)Arow[kk];
        a[2 * v + 1] = (_Float16)Arow[kk + 1];
        b[2 * v]     = (_Float16)Bcol[kk * 64];
        b[2 * v + 1] = (_Float16)Bcol[(kk + 1) * 64];
      }
      acc = wmma16(a, b, acc);
    }
    const float* Arow2 = Qc + (size_t)(tm * 16 + mn) * 64;
    const float* Bcol2 = KVpc + tn * 16 + mn;
    for (int k0 = 0; k0 < 64; k0 += 32) {
      v16h a, b;
#pragma unroll
      for (int v = 0; v < 8; ++v) {
        const int kk = k0 + frag_k(v, h);
        const float2 ta = *(const float2*)(Arow2 + kk);
        a[2 * v] = (_Float16)ta.x; a[2 * v + 1] = (_Float16)ta.y;
        b[2 * v]     = (_Float16)Bcol2[kk * 64];
        b[2 * v + 1] = (_Float16)Bcol2[(kk + 1) * 64];
      }
      acc = wmma16(a, b, acc);
    }
  }
#pragma unroll
  for (int v = 0; v < 8; ++v) {
    const int t = tm * 16 + v + 8 * h;
    Oc[t * 64 + tn * 16 + mn] = (_Float16)(acc[v] / den[t]);
  }
}

// ---------------------------------------------------------------------------
// LayerNorm: one block per row; writes f32 + optional f16 shadow.
// ---------------------------------------------------------------------------
__global__ void layernorm_kernel(const float* __restrict__ X,
                                 const float* __restrict__ g,
                                 const float* __restrict__ bta,
                                 float* __restrict__ Y,
                                 _Float16* __restrict__ Yh, int D) {
  const int row = blockIdx.x;
  const float* x = X + (size_t)row * D;
  float* y = Y + (size_t)row * D;
  _Float16* yh = Yh ? Yh + (size_t)row * D : nullptr;
  __shared__ float sh_s[8], sh_s2[8];
  const int wave = threadIdx.x >> 5, lane = threadIdx.x & 31;
  float s = 0.f, s2 = 0.f;
  for (int d = threadIdx.x; d < D; d += blockDim.x) {
    const float v = x[d];
    s += v; s2 += v * v;
  }
  for (int off = 16; off; off >>= 1) {
    s  += __shfl_xor(s, off, 32);
    s2 += __shfl_xor(s2, off, 32);
  }
  if (lane == 0) { sh_s[wave] = s; sh_s2[wave] = s2; }
  __syncthreads();
  if (threadIdx.x == 0) {
    float a = 0.f, b = 0.f;
    for (int i = 0; i < 8; ++i) { a += sh_s[i]; b += sh_s2[i]; }
    sh_s[0] = a; sh_s2[0] = b;
  }
  __syncthreads();
  const float mean = sh_s[0] / D;
  const float var  = sh_s2[0] / D - mean * mean;
  const float inv  = rsqrtf(var + 1e-5f);
  for (int d = threadIdx.x; d < D; d += blockDim.x) {
    const float v = (x[d] - mean) * inv * g[d] + bta[d];
    y[d] = v;
    if (yh) yh[d] = (_Float16)v;
  }
}

// ---------------------------------------------------------------------------
extern "C" void kernel_launch(void* const* d_in, const int* in_sizes, int n_in,
                              void* d_out, int out_size, void* d_ws, size_t ws_size,
                              hipStream_t stream) {
  (void)in_sizes; (void)n_in; (void)out_size; (void)ws_size;

  const int*   idx = (const int*)d_in[0];
  const float* TE  = (const float*)d_in[1];
  const float* PE  = (const float*)d_in[2];
  const float* Wq  = (const float*)d_in[3];
  const float* Wk  = (const float*)d_in[4];
  const float* Wv  = (const float*)d_in[5];
  const float* Wo  = (const float*)d_in[6];
  const float* ng  = (const float*)d_in[7];
  const float* nb  = (const float*)d_in[8];
  const float* ong = (const float*)d_in[9];
  const float* onb = (const float*)d_in[10];
  const float* W1  = (const float*)d_in[11];
  const float* b1  = (const float*)d_in[12];
  const float* W2  = (const float*)d_in[13];
  const float* b2  = (const float*)d_in[14];
  float* out = (float*)d_out;

  constexpr int Bb = 2, Ll = 2048, Dd = 1024, HIDd = 4096, Vv = 32000;
  constexpr int KDd = 64, NHh = 2, Tc = 64;
  const int BL = Bb * Ll;
  const int NC = Ll / Tc;

  char* p = (char*)d_ws;
  auto alloc_f = [&](size_t n) {
    float* r = (float*)p; p += (n * sizeof(float) + 15) & ~(size_t)15; return r;
  };
  auto alloc_h = [&](size_t n) {
    _Float16* r = (_Float16*)p;
    p += (n * sizeof(_Float16) + 15) & ~(size_t)15; return r;
  };

  float* x   = alloc_f((size_t)BL * Dd);
  float* xt  = alloc_f((size_t)BL * Dd);
  float* Qb  = alloc_f((size_t)BL * KDd);
  float* Kb  = alloc_f((size_t)BL * KDd);
  float* Vb  = alloc_f((size_t)BL * KDd);
  float* KVc = alloc_f((size_t)Bb * NC * KDd * KDd);
  float* KVp = alloc_f((size_t)Bb * NC * KDd * KDd);
  float* Ksc = alloc_f((size_t)Bb * NC * KDd);
  float* Ksp = alloc_f((size_t)Bb * NC * KDd);

  _Float16* xh  = alloc_h((size_t)BL * Dd);
  _Float16* Abh = alloc_h((size_t)BL * KDd);
  _Float16* Hb  = alloc_h((size_t)BL * HIDd);
  _Float16* Wqh = alloc_h((size_t)NHh * Dd * KDd);
  _Float16* Wkh = alloc_h((size_t)NHh * Dd * KDd);
  _Float16* Wvh = alloc_h((size_t)NHh * Dd * KDd);
  _Float16* Woh = alloc_h((size_t)NHh * KDd * Dd);
  _Float16* W1h = alloc_h((size_t)Dd * HIDd);
  _Float16* W2h = alloc_h((size_t)HIDd * Vv);

  auto cvt = [&](const float* s, _Float16* d, size_t n) {
    cvt_f16_kernel<<<(int)((n + 255) / 256), 256, 0, stream>>>(s, d, n);
  };
  cvt(Wq, Wqh, (size_t)NHh * Dd * KDd);
  cvt(Wk, Wkh, (size_t)NHh * Dd * KDd);
  cvt(Wv, Wvh, (size_t)NHh * Dd * KDd);
  cvt(Wo, Woh, (size_t)NHh * KDd * Dd);
  cvt(W1, W1h, (size_t)Dd * HIDd);
  cvt(W2, W2h, (size_t)HIDd * Vv);

  { // embedding (f32 + f16 shadow)
    const size_t total = (size_t)BL * Dd;
    embed_kernel<<<(int)((total + 255) / 256), 256, 0, stream>>>(idx, TE, PE, x,
                                                                 xh, BL, Ll, Dd);
  }

  auto gemmS = [&](const _Float16* A, const _Float16* Bm, float* C, int M,
                   int N, int K, int act) {
    const long tiles = (long)(M / 16) * (N / 16);
    wmma_gemm_small<<<(int)((tiles + 7) / 8), 256, 0, stream>>>(A, Bm, C, M, N,
                                                                K, act);
  };
  auto gemmT = [&](const _Float16* A, const _Float16* Bm, const float* bias,
                   const float* res, float* C, _Float16* C16, int M, int N,
                   int K, int act) {
    wmma_gemm_tiled<<<(M / TBM) * (N / TBN), 256, 0, stream>>>(
        A, Bm, bias, res, C, C16, M, N, K, act);
  };

  for (int i = 0; i < NHh; ++i) {
    gemmS(xh, Wqh + (size_t)i * Dd * KDd, Qb, BL, KDd, Dd, 1);
    gemmS(xh, Wkh + (size_t)i * Dd * KDd, Kb, BL, KDd, Dd, 1);
    gemmS(xh, Wvh + (size_t)i * Dd * KDd, Vb, BL, KDd, Dd, 0);
    chunk_kv_kernel<<<Bb * NC, 512, 0, stream>>>(Kb, Vb, KVc, Ksc);
    prefix_kernel<<<Bb, 256, 0, stream>>>(KVc, Ksc, KVp, Ksp, NC);
    attn_chunk_kernel<<<Bb * NC, 512, 0, stream>>>(Qb, Kb, Vb, KVp, Ksp, Abh);
    gemmT(Abh, Woh + (size_t)i * KDd * Dd, nullptr, x, xt, nullptr,
          BL, Dd, KDd, 0);
    layernorm_kernel<<<BL, 256, 0, stream>>>(xt, ng + (size_t)i * Dd,
                                             nb + (size_t)i * Dd, x, xh, Dd);
  }

  layernorm_kernel<<<BL, 256, 0, stream>>>(x, ong, onb, xt, xh, Dd);
  gemmT(xh, W1h, b1, nullptr, nullptr, Hb, BL, HIDd, Dd, 2);   // gelu -> f16
  gemmT(Hb, W2h, b2, nullptr, out, nullptr, BL, Vv, HIDd, 0);  // logits f32
}
